// GRUMessage_14422500180206
// MI455X (gfx1250) — compile-verified, matrix operand order checked by reading, not run on hardware
//
#include <hip/hip_runtime.h>
#include <hip/hip_bf16.h>
#include <math.h>

// ---------------------------------------------------------------------------
// Problem dimensions (match reference)
// ---------------------------------------------------------------------------
#define BB   16
#define RR   512
#define FF   2048
#define SS   81
#define II   1024
#define ROWS (BB * RR)   // 8192
#define F3   (3 * FF)    // 6144
#define N_ITERS 2

// GEMM tiling: 128x128 block tile, 4 waves (2x2), 64x64 per wave, K-tile 32.
#define TM  128
#define TN  128
#define TK  32
#define LDK 48           // padded LDS row stride in bf16 elems (96B, 16B-aligned)
#define GEMM_THREADS 128

typedef __bf16 bf16_t;
typedef __attribute__((ext_vector_type(16))) __bf16 v16bf;
typedef __attribute__((ext_vector_type(8)))  float  v8f;

union FragAB { v16bf v; uint4 q[2]; };
union FragC  { v8f   v; float f[8]; };
union Pack8  { uint4 q; bf16_t e[8]; };

// Load one 16x32 bf16 fragment from LDS laid out [16 rows][LDK k].
// ISA 16-bit fragment layout: lanes 0-15 hold K 0-7 / 16-23, lanes 16-31 hold
// K 8-15 / 24-31, row = lane & 15.  Two ds_load_b128 per lane.
__device__ __forceinline__ FragAB load_frag(const bf16_t* lds, int rowBase, int lane) {
  const bf16_t* p = lds + (rowBase + (lane & 15)) * LDK + ((lane & 16) ? 8 : 0);
  FragAB f;
  f.q[0] = *(const uint4*)(p);        // K chunk 0 (8 bf16 = 16B)
  f.q[1] = *(const uint4*)(p + 16);   // K chunk +16
  return f;
}

__device__ __forceinline__ Pack8 cvt8(float4 a0, float4 a1) {
  Pack8 u;
  u.e[0] = (bf16_t)a0.x; u.e[1] = (bf16_t)a0.y; u.e[2] = (bf16_t)a0.z; u.e[3] = (bf16_t)a0.w;
  u.e[4] = (bf16_t)a1.x; u.e[5] = (bf16_t)a1.y; u.e[6] = (bf16_t)a1.z; u.e[7] = (bf16_t)a1.w;
  return u;
}

// ---------------------------------------------------------------------------
// Generic bf16 WMMA GEMM:  C[M,N] = A[M,K] @ B + bias[N]
//   TRANSB == false : B is [K,N] row-major
//   TRANSB == true  : B is [N,K] row-major (torch weight layout, x @ W^T)
// fp32 operands converted to bf16 while staging to LDS; f32 accumulation.
// M must be a multiple of TM.  K and N tails handled branch-free (clamped
// addresses + selects) so the staging loop never scalarizes into exec-mask
// branches on the hot path.
// ---------------------------------------------------------------------------
template<bool TRANSB>
__global__ __launch_bounds__(GEMM_THREADS)
void gemm_bf16_wmma(const float* __restrict__ A, const float* __restrict__ Bm,
                    const float* __restrict__ bias, float* __restrict__ C,
                    int M, int N, int K) {
  __shared__ __align__(16) bf16_t lA[2][TM * LDK];   // 2 x 12KB
  __shared__ __align__(16) bf16_t lB[2][TN * LDK];   // 2 x 12KB

  const int tid  = threadIdx.x;     // 0..127
  const int lane = tid & 31;
  const int wave = tid >> 5;        // 0..3
  const int wm   = wave >> 1;       // 0..1 -> 64-row slab
  const int wn   = wave & 1;        // 0..1 -> 64-col slab
  const long m0  = (long)blockIdx.y * TM;
  const long n0  = (long)blockIdx.x * TN;

  FragC acc[4][4];
#pragma unroll
  for (int i = 0; i < 4; ++i)
#pragma unroll
    for (int j = 0; j < 4; ++j)
#pragma unroll
      for (int v = 0; v < 8; ++v) acc[i][j].f[v] = 0.0f;

  const bool aVec = ((K & 3) == 0);                 // row stride 16B-aligned
  const bool bVecT = ((K & 3) == 0);
  const bool bVecN = ((N & 3) == 0);

  // ---- staging: global fp32 -> bf16 -> LDS -------------------------------
  auto stage = [&](int buf, int kt) {
    // A tile: 128 rows x 32 k ; one row per thread (32 elems)
    {
      long gr = m0 + tid;
      bf16_t* dst = lA[buf] + tid * LDK;
      const float* src = A + gr * (long)K + kt;
      if (aVec && (kt + TK <= K)) {
#pragma unroll
        for (int c = 0; c < 4; ++c) {
          float4 a0 = *(const float4*)(src + c * 8);
          float4 a1 = *(const float4*)(src + c * 8 + 4);
          *(uint4*)(dst + c * 8) = cvt8(a0, a1).q;
        }
      } else {
        // clamped addresses + selects: no exec-mask branches
#pragma unroll
        for (int j = 0; j < 32; ++j) {
          int gk  = kt + j;
          int gkc = gk < K ? gk : K - 1;
          float x = A[gr * (long)K + gkc];
          dst[j] = (bf16_t)(gk < K ? x : 0.0f);
        }
      }
    }
    // B tile staged N-major: lB[n][k]
    if (TRANSB) {
      long gn = n0 + tid;           // one n-row per thread
      bf16_t* dst = lB[buf] + tid * LDK;
      const float* src = Bm + gn * (long)K + kt;
      if (bVecT && (kt + TK <= K) && (n0 + TN <= N)) {
#pragma unroll
        for (int c = 0; c < 4; ++c) {
          float4 a0 = *(const float4*)(src + c * 8);
          float4 a1 = *(const float4*)(src + c * 8 + 4);
          *(uint4*)(dst + c * 8) = cvt8(a0, a1).q;
        }
      } else {
        long gnc = gn < N ? gn : N - 1;
#pragma unroll
        for (int j = 0; j < 32; ++j) {
          int gk  = kt + j;
          int gkc = gk < K ? gk : K - 1;
          float x = Bm[gnc * (long)K + gkc];
          dst[j] = (bf16_t)((gk < K && gn < N) ? x : 0.0f);
        }
      }
    } else {
      int kk = tid >> 2;            // 0..31 : k within tile
      int nc = (tid & 3) * 32;      // 0..96 : n chunk of 32
      int gk = kt + kk;
      if (bVecN && (gk < K) && (n0 + TN <= N)) {
        const float* src = Bm + (long)gk * N + n0 + nc;
#pragma unroll
        for (int c = 0; c < 8; ++c) {
          float4 f = *(const float4*)(src + c * 4);
          lB[buf][(nc + c * 4 + 0) * LDK + kk] = (bf16_t)f.x;
          lB[buf][(nc + c * 4 + 1) * LDK + kk] = (bf16_t)f.y;
          lB[buf][(nc + c * 4 + 2) * LDK + kk] = (bf16_t)f.z;
          lB[buf][(nc + c * 4 + 3) * LDK + kk] = (bf16_t)f.w;
        }
      } else {
        int gkc = gk < K ? gk : K - 1;
#pragma unroll
        for (int j = 0; j < 32; ++j) {
          long gn  = n0 + nc + j;
          long gnc = gn < N ? gn : N - 1;
          float x = Bm[(long)gkc * N + gnc];
          lB[buf][(nc + j) * LDK + kk] = (bf16_t)((gk < K && gn < N) ? x : 0.0f);
        }
      }
    }
  };

  const int nTiles = (K + TK - 1) / TK;
  stage(0, 0);
  __syncthreads();

  for (int t = 0; t < nTiles; ++t) {
    const int buf = t & 1;

    // issue fragment loads for this tile first (overlap with next staging)
    FragAB a[4];
#pragma unroll
    for (int mi = 0; mi < 4; ++mi)
      a[mi] = load_frag(lA[buf], wm * 64 + mi * 16, lane);

    // prefetch/stage next K-tile into the other buffer
    if (t + 1 < nTiles) stage(buf ^ 1, (t + 1) * TK);

#pragma unroll
    for (int ni = 0; ni < 4; ++ni) {
      FragAB b = load_frag(lB[buf], wn * 64 + ni * 16, lane);
#pragma unroll
      for (int mi = 0; mi < 4; ++mi)
        acc[mi][ni].v = __builtin_amdgcn_wmma_f32_16x16x32_bf16(
            false, a[mi].v, false, b.v, (short)0, acc[mi][ni].v, false, false);
    }
    __syncthreads();   // buf fully consumed; safe to overwrite next round
  }

  // ---- epilogue: + bias[n], masked on N tail only (M % TM == 0) ----------
  const int mlo = (lane & 16) ? 8 : 0;  // C/D layout: VGPR v -> M = v + mlo
  const int nco = lane & 15;            // N = lane & 15
#pragma unroll
  for (int mi = 0; mi < 4; ++mi) {
    long mbase = m0 + wm * 64 + mi * 16 + mlo;
#pragma unroll
    for (int ni = 0; ni < 4; ++ni) {
      long ncol = n0 + wn * 64 + ni * 16 + nco;
      if (ncol < N) {
        float bv = bias[ncol];
        float* cp = C + mbase * N + ncol;
#pragma unroll
        for (int v = 0; v < 8; ++v) cp[(long)v * N] = acc[mi][ni].f[v] + bv;
      }
    }
  }
}

// ---------------------------------------------------------------------------
// Elementwise / small kernels (HBM-bound, VALU)
// ---------------------------------------------------------------------------

// g <- relu(h * g * (b @ W_box + b_box)) ; g holds s@W_score+b_score on entry
__global__ __launch_bounds__(256)
void gate_kernel(const float* __restrict__ h, float* __restrict__ g,
                 const float* __restrict__ bx, const float* __restrict__ W_box,
                 const float* __restrict__ b_box) {
  long idx = (long)blockIdx.x * 256 + threadIdx.x;  // ROWS*FF
  long row = idx / FF;
  int  f   = (int)(idx - row * FF);
  const float* b4 = bx + row * 4;
  float bp = b_box[f];
  bp += b4[0] * W_box[0 * FF + f];
  bp += b4[1] * W_box[1 * FF + f];
  bp += b4[2] * W_box[2 * FF + f];
  bp += b4[3] * W_box[3 * FF + f];
  float x = h[idx] * g[idx] * bp;
  g[idx] = x > 0.0f ? x : 0.0f;
}

// msum[b,i] = sum_r m[b,r,i]
__global__ __launch_bounds__(256)
void colsum_kernel(const float* __restrict__ m, float* __restrict__ msum) {
  int idx = blockIdx.x * 256 + threadIdx.x;   // BB*II
  int b = idx / II, i = idx - (idx / II) * II;
  const float* p = m + (long)b * RR * II + i;
  float s = 0.0f;
  for (int r = 0; r < RR; ++r) s += p[(long)r * II];
  msum[idx] = s;
}

// m <- (msum[b] - m) / (R-1)   (leave-one-out mean, in place)
__global__ __launch_bounds__(256)
void loo_kernel(float* __restrict__ m, const float* __restrict__ msum, float invDenom) {
  long idx = (long)blockIdx.x * 256 + threadIdx.x;  // ROWS*II
  long b = idx / ((long)RR * II);
  int  i = (int)(idx % II);
  m[idx] = (msum[b * II + i] - m[idx]) * invDenom;
}

// torch GRUCell combine: gates stacked [r, z, n]; h updated in place
__global__ __launch_bounds__(256)
void gru_kernel(const float* __restrict__ gi, const float* __restrict__ gh,
                float* __restrict__ h) {
  long idx = (long)blockIdx.x * 256 + threadIdx.x;  // ROWS*FF
  long row = idx / FF;
  int  f   = (int)(idx - row * FF);
  const float* gir = gi + row * (long)F3;
  const float* ghr = gh + row * (long)F3;
  float r = 1.0f / (1.0f + __expf(-(gir[f]          + ghr[f])));
  float z = 1.0f / (1.0f + __expf(-(gir[FF + f]     + ghr[FF + f])));
  float n = tanhf(gir[2 * FF + f] + r * ghr[2 * FF + f]);
  h[idx] = (1.0f - z) * n + z * h[idx];
}

// softmax over S=81 classes, one 128-thread block (4 waves) per row, in place
__global__ __launch_bounds__(128)
void softmax_kernel(float* __restrict__ s) {
  __shared__ float red[4];
  int row  = blockIdx.x;
  int t    = threadIdx.x;
  int lane = t & 31, wv = t >> 5;
  float* p = s + (long)row * SS;
  float v = (t < SS) ? p[t] : -3.4e38f;
#pragma unroll
  for (int off = 16; off > 0; off >>= 1) v = fmaxf(v, __shfl_xor(v, off, 32));
  if (lane == 0) red[wv] = v;
  __syncthreads();
  float mx = fmaxf(fmaxf(red[0], red[1]), fmaxf(red[2], red[3]));
  __syncthreads();
  float e = (t < SS) ? __expf(p[t] - mx) : 0.0f;
  float sum = e;
#pragma unroll
  for (int off = 16; off > 0; off >>= 1) sum += __shfl_xor(sum, off, 32);
  if (lane == 0) red[wv] = sum;
  __syncthreads();
  float tot = red[0] + red[1] + red[2] + red[3];
  if (t < SS) p[t] = e / tot;
}

// out[row, 0..3] = h[row] @ W_bb + b_bb   (N=4: one wave per row, shuffle-reduce)
__global__ __launch_bounds__(256)
void boxproj_kernel(const float* __restrict__ h, const float* __restrict__ Wbb,
                    const float* __restrict__ bbb, float* __restrict__ out) {
  int row  = blockIdx.x * 8 + (threadIdx.x >> 5);
  int lane = threadIdx.x & 31;
  const float* hr = h + (long)row * FF;
  float a0 = 0.f, a1 = 0.f, a2 = 0.f, a3 = 0.f;
  for (int k = lane; k < FF; k += 32) {
    float hv = hr[k];
    const float* w = Wbb + k * 4;
    a0 += hv * w[0]; a1 += hv * w[1]; a2 += hv * w[2]; a3 += hv * w[3];
  }
#pragma unroll
  for (int off = 16; off > 0; off >>= 1) {
    a0 += __shfl_xor(a0, off, 32);
    a1 += __shfl_xor(a1, off, 32);
    a2 += __shfl_xor(a2, off, 32);
    a3 += __shfl_xor(a3, off, 32);
  }
  if (lane == 0) {
    float* o = out + (long)row * 4;
    o[0] = a0 + bbb[0]; o[1] = a1 + bbb[1];
    o[2] = a2 + bbb[2]; o[3] = a3 + bbb[3];
  }
}

// ---------------------------------------------------------------------------
// Host orchestration
// ---------------------------------------------------------------------------
extern "C" void kernel_launch(void* const* d_in, const int* in_sizes, int n_in,
                              void* d_out, int out_size, void* d_ws, size_t ws_size,
                              hipStream_t stream) {
  const float* features = (const float*)d_in[0];
  const float* scores   = (const float*)d_in[1];
  const float* boxes    = (const float*)d_in[2];
  const float* W_score  = (const float*)d_in[3];
  const float* b_score  = (const float*)d_in[4];
  const float* W_box    = (const float*)d_in[5];
  const float* b_box    = (const float*)d_in[6];
  const float* W_in     = (const float*)d_in[7];
  const float* b_in     = (const float*)d_in[8];
  const float* W_ih     = (const float*)d_in[9];
  const float* b_ih     = (const float*)d_in[10];
  const float* W_hh     = (const float*)d_in[11];
  const float* b_hh     = (const float*)d_in[12];
  const float* W_cls    = (const float*)d_in[13];
  const float* b_cls    = (const float*)d_in[14];
  const float* W_bb     = (const float*)d_in[15];
  const float* b_bb     = (const float*)d_in[16];

  // workspace layout (floats)
  float* ws   = (float*)d_ws;
  float* h    = ws;                             // ROWS*FF
  float* g    = h    + (size_t)ROWS * FF;       // ROWS*FF (proj_s, then gated)
  float* m    = g    + (size_t)ROWS * FF;       // ROWS*II (message, then inp)
  float* msum = m    + (size_t)ROWS * II;       // BB*II
  float* gi   = msum + (size_t)BB * II;         // ROWS*F3
  float* gh   = gi   + (size_t)ROWS * F3;       // ROWS*F3
  float* sb   = gh   + (size_t)ROWS * F3;       // ROWS*SS (scores / logits)
  float* bx   = sb   + (size_t)ROWS * SS;       // ROWS*4

  // initialize state from inputs (never mutate d_in)
  hipMemcpyAsync(h,  features, sizeof(float) * (size_t)ROWS * FF, hipMemcpyDeviceToDevice, stream);
  hipMemcpyAsync(sb, scores,   sizeof(float) * (size_t)ROWS * SS, hipMemcpyDeviceToDevice, stream);
  hipMemcpyAsync(bx, boxes,    sizeof(float) * (size_t)ROWS * 4,  hipMemcpyDeviceToDevice, stream);

  const float invDenom = 1.0f / (float)(RR - 1);
  dim3 gblk(GEMM_THREADS);
  dim3 eblk(256);

  for (int it = 0; it < N_ITERS; ++it) {
    // g = s @ W_score + b_score                    [8192 x 2048, K=81]
    gemm_bf16_wmma<false><<<dim3(FF / TN, ROWS / TM), gblk, 0, stream>>>(
        sb, W_score, b_score, g, ROWS, FF, SS);
    // g = relu(h * g * (b @ W_box + b_box))
    gate_kernel<<<(unsigned)(((long)ROWS * FF) / 256), eblk, 0, stream>>>(
        h, g, bx, W_box, b_box);
    // m = g @ W_in + b_in                          [8192 x 1024, K=2048]
    gemm_bf16_wmma<false><<<dim3(II / TN, ROWS / TM), gblk, 0, stream>>>(
        g, W_in, b_in, m, ROWS, II, FF);
    // leave-one-out mean over rois per image
    colsum_kernel<<<(BB * II) / 256, eblk, 0, stream>>>(m, msum);
    loo_kernel<<<(unsigned)(((long)ROWS * II) / 256), eblk, 0, stream>>>(m, msum, invDenom);
    // gi = inp @ W_ih^T + b_ih                     [8192 x 6144, K=1024]
    gemm_bf16_wmma<true><<<dim3(F3 / TN, ROWS / TM), gblk, 0, stream>>>(
        m, W_ih, b_ih, gi, ROWS, F3, II);
    // gh = h @ W_hh^T + b_hh                       [8192 x 6144, K=2048]
    gemm_bf16_wmma<true><<<dim3(F3 / TN, ROWS / TM), gblk, 0, stream>>>(
        h, W_hh, b_hh, gh, ROWS, F3, FF);
    // h = GRU(gi, gh, h)
    gru_kernel<<<(unsigned)(((long)ROWS * FF) / 256), eblk, 0, stream>>>(gi, gh, h);
    // sb = softmax(h @ W_cls + b_cls)              [8192 x 81, K=2048]
    gemm_bf16_wmma<false><<<dim3((SS + TN - 1) / TN, ROWS / TM), gblk, 0, stream>>>(
        h, W_cls, b_cls, sb, ROWS, SS, FF);
    softmax_kernel<<<ROWS, dim3(128), 0, stream>>>(sb);
    // bx = h @ W_bb + b_bb                         [8192 x 4, K=2048]
    boxproj_kernel<<<ROWS / 8, eblk, 0, stream>>>(h, W_bb, b_bb, bx);
  }

  // outputs: new_scores [ROWS,SS] then new_boxes [ROWS,4], concatenated flat
  hipMemcpyAsync(d_out, sb, sizeof(float) * (size_t)ROWS * SS,
                 hipMemcpyDeviceToDevice, stream);
  hipMemcpyAsync((float*)d_out + (size_t)ROWS * SS, bx, sizeof(float) * (size_t)ROWS * 4,
                 hipMemcpyDeviceToDevice, stream);
}
